// TopKSAE_22565758173710
// MI455X (gfx1250) — matrix operand order repaired and checked by reading, not compiled, not run on hardware
//
#include <hip/hip_runtime.h>

// ---------------- problem constants (from reference) ----------------
#define B_ROWS 65536
#define D_IN   640
#define D_LAT  2560
#define TOPK   32
#define NKSTEP (D_IN / 32)  // 20
#define TILE_BYTES 163840   // one K-step B tile: 2560 rows * 32 k * 2B (=16*D_LAT*4 too)

// ---------------- CDNA5 vector types ----------------
typedef __bf16 bf16;
typedef __attribute__((ext_vector_type(16))) __bf16 v16bf;
typedef __attribute__((ext_vector_type(8)))  __bf16 v8bf;
typedef __attribute__((ext_vector_type(8)))  float  v8f;
typedef __attribute__((ext_vector_type(4)))  float  v4f;
typedef __attribute__((ext_vector_type(4)))  unsigned v4u;
typedef __attribute__((ext_vector_type(8)))  int    v8i;
typedef __attribute__((ext_vector_type(4)))  int    v4i;

// ============================================================================
// Kernel 1: fp32 -> bf16 (round-to-nearest-even bit trick).
// ============================================================================
__global__ void cvt_f32_bf16_kernel(const float* __restrict__ in,
                                    unsigned short* __restrict__ out, int n) {
    int i = blockIdx.x * blockDim.x + threadIdx.x;
    if (i < n) {
        union { float f; unsigned u; } c;
        c.f = in[i];
        unsigned u = c.u;
        unsigned r = (u + 0x7fffu + ((u >> 16) & 1u)) >> 16;
        out[i] = (unsigned short)r;
    }
}

// ============================================================================
// Kernel 2: transpose W_dec [D_IN, D_LAT] -> W_decT [D_LAT, D_IN] (fp32).
// ============================================================================
__global__ void transpose_wdec_kernel(const float* __restrict__ in,
                                      float* __restrict__ out) {
    int gid = blockIdx.x * blockDim.x + threadIdx.x;
    if (gid < D_LAT * D_IN) {
        int l = gid / D_IN;
        int d = gid - l * D_IN;
        out[gid] = in[d * D_LAT + l];
    }
}

// ============================================================================
// TDM: one descriptor moves the whole 2560x32 bf16 B-tile (160 KB) for one
// K-step into LDS.  2D tensor, data_size=2B, dim0 (contiguous K) len 640,
// dim0 stride 640, dim1 (latent rows) len 2560; tile = 32 x 2560.
// LDS layout after the copy: byte(l, k) = l*64 + (k-k0)*2.
// ============================================================================
__device__ __forceinline__ void tdm_load_wenc_tile(const bf16* gtile,
                                                   unsigned lds_byte_off) {
    unsigned long long ga = (unsigned long long)gtile;
    v4u g0;
    g0[0] = 1u;                                   // count=1, user descriptor
    g0[1] = lds_byte_off;                         // LDS destination (bytes)
    g0[2] = (unsigned)ga;                         // global_addr[31:0]
    g0[3] = (unsigned)((ga >> 32) & 0x01FFFFFFu)  // global_addr[56:32]
            | (2u << 30);                         // type = 2 ("image")
    v8i g1;
    g1[0] = (int)(1u << 16);                      // workgroup_mask=0, data_size=1 (2B)
    g1[1] = (int)((unsigned)(D_IN & 0xFFFF) << 16);   // tensor_dim0[15:0] = 640
    g1[2] = (int)((unsigned)(D_LAT & 0xFFFF) << 16);  // dim0 hi=0 | tensor_dim1 = 2560
    g1[3] = (int)(32u << 16);                     // tensor_dim1 hi=0 | tile_dim0 = 32
    g1[4] = D_LAT;                                // tile_dim1 = 2560 (tile_dim2 = 0)
    g1[5] = D_IN;                                 // tensor_dim0_stride = 640
    g1[6] = 0;
    g1[7] = 0;
    v4i z4 = {0, 0, 0, 0};                        // groups 2/3 unused (2D tensor)
    v8i z8 = {0, 0, 0, 0, 0, 0, 0, 0};
    __builtin_amdgcn_tensor_load_to_lds(g0, g1, z4, z4, z8, 0);
}

// ============================================================================
// Kernel 3: fused encoder GEMM (bf16 WMMA, fp32 accum) + bias + top-32 +
// sparse scatter.  One block = 16 rows of x, full D_LAT columns.
// B tiles arrive via TDM, double-buffered in 2x160 KB LDS (full 320 KB WGP
// LDS); buffer 0 is reused for the fp32 latent tile after the K loop.
// Per K-step: all 20 B fragments are loaded into distinct registers first,
// then the 20-wmma chain drains them (hides LDS latency); A(kk+1) HBM loads
// are issued under the chain of kk.
// ============================================================================
__global__ void __launch_bounds__(256)
encoder_topk_kernel(const bf16* __restrict__ xb,     // [B, D_IN] bf16
                    const bf16* __restrict__ wencb,  // [D_LAT, D_IN] bf16
                    const float* __restrict__ b_enc, // [D_LAT]
                    float* __restrict__ sparse_out,  // [B, D_LAT]
                    float* __restrict__ tk_val,      // [B, TOPK]
                    int*   __restrict__ tk_idx) {    // [B, TOPK]
    extern __shared__ float lat[];    // 2 * TILE_BYTES; floats view = buffer 0
    const int tid  = threadIdx.x;
    const int wave = tid >> 5;
    const int lane = tid & 31;
    const int lm   = lane & 15;       // M row (A/C) or N col (B/C)
    const int kh   = lane >> 4;       // K half selector
    const int m0   = blockIdx.x * 16;
    const int nbase = wave * 320;

    // LDS byte offset of the dynamic shared block (for the TDM descriptor)
    const unsigned lds_base =
        (unsigned)(unsigned long long)
            (__attribute__((address_space(3))) char*)(void*)lat;

    v8f acc[20];
    {
        v8f z = {0.f, 0.f, 0.f, 0.f, 0.f, 0.f, 0.f, 0.f};
        #pragma unroll
        for (int t = 0; t < 20; ++t) acc[t] = z;
    }

    const bf16* xrow = xb + (size_t)(m0 + lm) * D_IN;
    const char* smem_c = (const char*)lat;

    // prologue: kick off B tile 0 (TDM) and A fragment 0 (HBM)
    if (wave == 0) tdm_load_wenc_tile(wencb, lds_base);
    v8bf alo = *(const v8bf*)(xrow + kh * 16);
    v8bf ahi = *(const v8bf*)(xrow + kh * 16 + 8);

    #pragma unroll 1
    for (int kk = 0; kk < NKSTEP; ++kk) {
        if (wave == 0) {
            if (kk + 1 < NKSTEP) {
                // prefetch B tile kk+1 into the other buffer, then wait for
                // the oldest in-flight TDM op (tile kk) to land.
                tdm_load_wenc_tile(wencb + (kk + 1) * 32,
                                   lds_base + (unsigned)((kk + 1) & 1) * TILE_BYTES);
                __builtin_amdgcn_s_wait_tensorcnt(1);
            } else {
                __builtin_amdgcn_s_wait_tensorcnt(0);
            }
        }
        __syncthreads();   // tile kk visible to all waves

        // ---- pack A fragment for kk, then issue A loads for kk+1 ----
        v16bf a;
        #pragma unroll
        for (int e = 0; e < 8; ++e) { a[e] = alo[e]; a[e + 8] = ahi[e]; }
        if (kk + 1 < NKSTEP) {
            const int kn = (kk + 1) * 32;
            alo = *(const v8bf*)(xrow + kn + kh * 16);
            ahi = *(const v8bf*)(xrow + kn + kh * 16 + 8);
        }

        // ---- load ALL 20 B fragments into distinct registers, then drain
        //      the wmma chain (LDS latency hidden under the chain) ----
        const char* fbase =
            smem_c + (size_t)(kk & 1) * TILE_BYTES + (nbase + lm) * 64 + kh * 32;
        v16bf bfrag[20];
        #pragma unroll
        for (int t = 0; t < 20; ++t)
            bfrag[t] = *(const v16bf*)(fbase + t * 1024);
        #pragma unroll
        for (int t = 0; t < 20; ++t)
            acc[t] = __builtin_amdgcn_wmma_f32_16x16x32_bf16(
                false, a, false, bfrag[t], (short)0, acc[t], false, false);

        __syncthreads();   // done reading buf[kk&1]; safe to overwrite at kk+2
    }

    // ---- epilogue: bias + store latent tile to LDS buffer 0 ----
    #pragma unroll
    for (int t = 0; t < 20; ++t) {
        const int n = nbase + t * 16 + lm;
        const float bias = b_enc[n];
        #pragma unroll
        for (int r = 0; r < 8; ++r) {
            const int M = r + 8 * kh;
            lat[M * D_LAT + n] = acc[t][r] + bias;
        }
    }
    __syncthreads();

    // ---- per-wave top-32 over full rows; wave w handles rows 2w, 2w+1 ----
    for (int rr = 0; rr < 2; ++rr) {
        const int row = wave * 2 + rr;
        const int b   = m0 + row;
        const float* lrow = lat + row * D_LAT;

        unsigned long long ex0 = 0ull;   // per-lane exclusion bitmap (80 slots)
        unsigned ex1 = 0u;
        float myv = 0.f;
        int   myi = -1;

        for (int r = 0; r < TOPK; ++r) {
            float v = -__builtin_inff();
            int   ji = -1;
            #pragma unroll 1
            for (int s = 0; s < 80; ++s) {
                const bool ex = (s < 64) ? ((ex0 >> s) & 1ull)
                                         : ((ex1 >> (s - 64)) & 1u);
                const float t = lrow[lane + s * 32];
                if (!ex && t > v) { v = t; ji = lane + s * 32; }
            }
            #pragma unroll
            for (int off = 16; off >= 1; off >>= 1) {   // wave32 argmax
                const float ov = __shfl_xor(v, off, 32);
                const int   oi = __shfl_xor(ji, off, 32);
                if (ov > v || (ov == v && oi >= 0 && (ji < 0 || oi < ji))) {
                    v = ov; ji = oi;
                }
            }
            if (lane == r) { myv = v; myi = ji; }
            if (ji >= 0 && (ji & 31) == lane) {
                const int s = ji >> 5;
                if (s < 64) ex0 |= 1ull << s;
                else        ex1 |= 1u << (s - 64);
            }
        }

        float* orow = sparse_out + (size_t)b * D_LAT;
        const v4f z4 = {0.f, 0.f, 0.f, 0.f};
        #pragma unroll
        for (int j = lane; j < D_LAT / 4; j += 32) ((v4f*)orow)[j] = z4;
        if (myi >= 0) orow[myi] = myv;
        tk_val[b * TOPK + lane] = myv;
        tk_idx[b * TOPK + lane] = myi;
    }
}

// ============================================================================
// Kernel 4: sparse decoder. One wave per sample; 32 gathered rows of W_decT
// (L2-resident), v4f-vectorized (5x global_load_b128 per row), with
// global_prefetch on the next row.
// ============================================================================
__global__ void __launch_bounds__(256)
decoder_kernel(const float* __restrict__ wdecT,  // [D_LAT, D_IN]
               const float* __restrict__ b_dec,  // [D_IN]
               const float* __restrict__ tk_val,
               const int*   __restrict__ tk_idx,
               float* __restrict__ recon) {      // [B, D_IN]
    const int wave = threadIdx.x >> 5;
    const int lane = threadIdx.x & 31;
    const int b = blockIdx.x * 8 + wave;

    v4f acc[5];                       // 20 floats = 5 x v4f per lane
    const v4f* bd4 = (const v4f*)b_dec;
    #pragma unroll
    for (int j = 0; j < 5; ++j) acc[j] = bd4[lane + 32 * j];

    const float* tv = tk_val + b * TOPK;
    const int*   ti = tk_idx + b * TOPK;

    #pragma unroll 1
    for (int k = 0; k < TOPK; ++k) {
        const int   idx = ti[k];
        const float v   = tv[k];
        if (k + 1 < TOPK) {
            const int nidx = ti[k + 1];
            __builtin_prefetch(wdecT + (size_t)(nidx < 0 ? 0 : nidx) * D_IN,
                               0, 1);            // -> global_prefetch_b8
        }
        if (idx >= 0) {
            const v4f* w4 = (const v4f*)(wdecT + (size_t)idx * D_IN);
            #pragma unroll
            for (int j = 0; j < 5; ++j) acc[j] += v * w4[lane + 32 * j];
        }
    }
    v4f* out4 = (v4f*)(recon + (size_t)b * D_IN);
    #pragma unroll
    for (int j = 0; j < 5; ++j) out4[lane + 32 * j] = acc[j];
}

// ============================================================================
// Launch
// ============================================================================
extern "C" void kernel_launch(void* const* d_in, const int* in_sizes, int n_in,
                              void* d_out, int out_size, void* d_ws,
                              size_t ws_size, hipStream_t stream) {
    (void)in_sizes; (void)n_in; (void)out_size; (void)ws_size;
    const float* x     = (const float*)d_in[0];
    const float* W_enc = (const float*)d_in[1];
    const float* b_enc = (const float*)d_in[2];
    const float* W_dec = (const float*)d_in[3];
    const float* b_dec = (const float*)d_in[4];

    float* recon  = (float*)d_out;                          // [B, D_IN]
    float* sparse = (float*)d_out + (size_t)B_ROWS * D_IN;  // [B, D_LAT]

    char* ws = (char*)d_ws;
    unsigned short* xb  = (unsigned short*)ws; ws += (size_t)B_ROWS * D_IN * 2;
    unsigned short* web = (unsigned short*)ws; ws += (size_t)D_LAT * D_IN * 2;
    float* wdT = (float*)ws;                   ws += (size_t)D_LAT * D_IN * 4;
    float* tkv = (float*)ws;                   ws += (size_t)B_ROWS * TOPK * 4;
    int*   tki = (int*)ws;

    {
        int n = B_ROWS * D_IN;
        cvt_f32_bf16_kernel<<<(n + 255) / 256, 256, 0, stream>>>(x, xb, n);
    }
    {
        int n = D_LAT * D_IN;
        cvt_f32_bf16_kernel<<<(n + 255) / 256, 256, 0, stream>>>(W_enc, web, n);
    }
    transpose_wdec_kernel<<<(D_LAT * D_IN + 255) / 256, 256, 0, stream>>>(
        W_dec, wdT);

    const size_t smem = 2 * (size_t)TILE_BYTES;   // 320 KB: double-buffered TDM
    encoder_topk_kernel<<<B_ROWS / 16, 256, smem, stream>>>(
        (const bf16*)xb, (const bf16*)web, b_enc, sparse, tkv, tki);

    decoder_kernel<<<B_ROWS / 8, 256, 0, stream>>>(wdT, b_dec, tkv, tki, recon);
}
